// MLA_1331439862383
// MI455X (gfx1250) — matrix operand order repaired and checked by reading, not compiled, not run on hardware
//
#include <hip/hip_runtime.h>

// ---------------- types ----------------
typedef __bf16 bf16;
typedef __attribute__((ext_vector_type(16))) __bf16 v16bf;
typedef __attribute__((ext_vector_type(8)))  __bf16 bf16x8;
typedef __attribute__((ext_vector_type(8)))  float  v8f;
typedef __attribute__((ext_vector_type(4)))  int    v4i;

union FragU { v16bf v; bf16x8 h[2]; };

// ---- gfx1250 async global->LDS path (ASYNCcnt-tracked), with sync fallback ----
#if defined(__gfx1250__) && \
    __has_builtin(__builtin_amdgcn_global_load_async_to_lds_b128) && \
    __has_builtin(__builtin_amdgcn_s_wait_asynccnt)
#define USE_ASYNC_LDS 1
#endif

__device__ __forceinline__ void copy16_g2l(const bf16* g, bf16* l) {
#ifdef USE_ASYNC_LDS
  // builtin signature (probed from diagnostics): (AS1 v4i*, AS3 v4i*, imm off, imm cpol)
  __builtin_amdgcn_global_load_async_to_lds_b128(
      (__attribute__((address_space(1))) v4i*)(v4i*)g,
      (__attribute__((address_space(3))) v4i*)(v4i*)l, 0, 0);
#else
  *(bf16x8*)l = *(const bf16x8*)g;
#endif
}

__device__ __forceinline__ void async_join() {
#ifdef USE_ASYNC_LDS
  __builtin_amdgcn_s_wait_asynccnt(0);
#endif
}

__device__ __forceinline__ v8f vzero() {
  v8f z;
#pragma unroll
  for (int i = 0; i < 8; ++i) z[i] = 0.0f;
  return z;
}

// D = A(16x32) * B(32x16) + C, bf16 inputs, f32 accumulate
__device__ __forceinline__ v8f wmma_bf16(v16bf a, v16bf b, v8f c) {
  return __builtin_amdgcn_wmma_f32_16x16x32_bf16(false, a, false, b, (short)0, c,
                                                 false, false);
}

// A-fragment (16x32, M x K): lane L<16 holds row L, K = {0..7, 16..23};
// lane L>=16 holds row L-16, K = {8..15, 24..31}. Two 16B LDS loads.
__device__ __forceinline__ v16bf load_a_frag(const bf16* smem, int stride,
                                             int rowBase, int k0, int lane) {
  const int r  = lane & 15;
  const int hi = lane >> 4;
  const bf16* p = smem + (rowBase + r) * stride + k0;
  FragU f;
  f.h[0] = *(const bf16x8*)(p + 8 * hi);
  f.h[1] = *(const bf16x8*)(p + 16 + 8 * hi);
  return f.v;
}

// B-fragment (32x16, K x N), stored in LDS as Bt[col][k] (contraction contiguous):
// lane L<16 holds col L, K = {0..7, 16..23}; lane L>=16 holds col L-16, K = {8..15, 24..31}.
__device__ __forceinline__ v16bf load_b_frag(const bf16* smem, int stride,
                                             int colBase, int k0, int lane) {
  const int c  = lane & 15;
  const int hi = lane >> 4;
  const bf16* p = smem + (colBase + c) * stride + k0;
  FragU f;
  f.h[0] = *(const bf16x8*)(p + 8 * hi);
  f.h[1] = *(const bf16x8*)(p + 16 + 8 * hi);
  return f.v;
}

// ---------------- elementwise prep kernels ----------------
__global__ __launch_bounds__(256) void cvt_f32_bf16(const float* __restrict__ in,
                                                    bf16* __restrict__ out,
                                                    long long n) {
  long long i = (long long)blockIdx.x * 256 + threadIdx.x;
  if (i < n) out[i] = (bf16)in[i];
}

// Wt[n][k] = (bf16) W[k][n]
__global__ __launch_bounds__(256) void transpose_cvt(const float* __restrict__ W,
                                                     bf16* __restrict__ Wt,
                                                     int K, int N) {
  long long i = (long long)blockIdx.x * 256 + threadIdx.x;
  if (i >= (long long)K * N) return;
  int n = (int)(i % N);
  long long k = i / N;
  Wt[(long long)n * K + k] = (bf16)W[i];
}

// ---------------- generic bf16 WMMA GEMM ----------------
// C[M][N] = A[M][K] @ Bt[N][K]^T ; 128x128 block tile, 8 waves of 32.
// Output mode is a compile-time template -> branch-free epilogue.
template <bool STORE_F, bool STORE_B>
__global__ __launch_bounds__(256) void gemm_bf16_wmma(const bf16* __restrict__ A,
                                                      const bf16* __restrict__ Bt,
                                                      float* __restrict__ outF,
                                                      bf16* __restrict__ outB,
                                                      int M, int N, int K) {
  __shared__ bf16 As[128 * 48];
  __shared__ bf16 Bs[128 * 48];

  const int tid  = threadIdx.x;
  const int lane = tid & 31;
  const int wid  = tid >> 5;
  const int m0   = (wid & 3) * 32;   // wave grid 4 (M) x 2 (N)
  const int n0   = (wid >> 2) * 64;
  const long long mBase = (long long)blockIdx.y * 128;
  const long long nBase = (long long)blockIdx.x * 128;

  v8f acc[2][4];
#pragma unroll
  for (int i = 0; i < 2; ++i)
#pragma unroll
    for (int j = 0; j < 4; ++j) acc[i][j] = vzero();

  const int fr = tid >> 1;           // fill row 0..127
  const int fc = (tid & 1) * 16;     // fill col 0 / 16
  const bf16* gA = A  + (mBase + fr) * (long long)K + fc;
  const bf16* gB = Bt + (nBase + fr) * (long long)K + fc;

  for (int k0 = 0; k0 < K; k0 += 32) {
    __syncthreads();
    copy16_g2l(gA + k0,     &As[fr * 48 + fc]);
    copy16_g2l(gA + k0 + 8, &As[fr * 48 + fc + 8]);
    copy16_g2l(gB + k0,     &Bs[fr * 48 + fc]);
    copy16_g2l(gB + k0 + 8, &Bs[fr * 48 + fc + 8]);
    __builtin_prefetch(gA + k0 + 32, 0, 0);   // global_prefetch_b8
    __builtin_prefetch(gB + k0 + 32, 0, 0);
    async_join();
    __syncthreads();

    v16bf af[2], bfg[4];
#pragma unroll
    for (int i = 0; i < 2; ++i) af[i]  = load_a_frag(As, 48, m0 + 16 * i, 0, lane);
#pragma unroll
    for (int j = 0; j < 4; ++j) bfg[j] = load_b_frag(Bs, 48, n0 + 16 * j, 0, lane);
#pragma unroll
    for (int i = 0; i < 2; ++i)
#pragma unroll
      for (int j = 0; j < 4; ++j) acc[i][j] = wmma_bf16(af[i], bfg[j], acc[i][j]);
  }

  const int col = lane & 15;
  const int hi  = lane >> 4;
#pragma unroll
  for (int i = 0; i < 2; ++i)
#pragma unroll
    for (int j = 0; j < 4; ++j) {
      long long rBase = mBase + m0 + 16 * i + 8 * hi;
      long long cIdx  = nBase + n0 + 16 * j + col;
#pragma unroll
      for (int e = 0; e < 8; ++e) {
        float v = acc[i][j][e];
        long long o = (rBase + e) * (long long)N + cIdx;
        if constexpr (STORE_F) outF[o] = v;
        if constexpr (STORE_B) outB[o] = (bf16)v;
      }
    }
}

// ---------------- RoPE + K/Q packing ----------------
// K/Q buffers: (B,S,nh,128) bf16 = [k_c | rope(k_r)] per head.
// Also writes fp32 k_rope (B,S,nh,64) to d_out region 3.
__global__ __launch_bounds__(256) void rope_pack_kernel(
    const float* __restrict__ kr, const float* __restrict__ qr,
    const bf16* __restrict__ kc, const bf16* __restrict__ qc,
    bf16* __restrict__ Kb, bf16* __restrict__ Qb, float* __restrict__ krope) {
  long long idx = (long long)blockIdx.x * 256 + threadIdx.x;
  if (idx >= 4LL * 1024 * 16 * 64) return;
  const int d = (int)(idx & 63);
  long long bsh = idx >> 6;          // (b*S+s)*16 + h
  const int h = (int)(bsh & 15);
  long long bs = bsh >> 4;           // b*S + s
  const int s = (int)(bs & 1023);

  const long long lowBase = bs * 1024 + (long long)h * 64;
  const long long outBase = bs * 2048 + (long long)h * 128;

  // compressed halves (dims 0..63)
  Kb[outBase + d] = kc[lowBase + d];
  Qb[outBase + d] = qc[lowBase + d];

  float kv, qv;
  if (d < 32) {
    const int j = d & 15;                 // rotation = concat(freqs, freqs)
    float inv = __powf(10000.0f, -(float)(2 * j) / 32.0f);
    float fr  = ((float)s / 40.0f) * inv; // ROPE_SCALE = 40
    float c = __cosf(fr), sn = __sinf(fr);
    int pair = (d < 16) ? d + 16 : d - 16;
    float sg = (d < 16) ? -1.0f : 1.0f;
    kv = kr[lowBase + d] * c + sg * kr[lowBase + pair] * sn;
    qv = qr[lowBase + d] * c + sg * qr[lowBase + pair] * sn;
  } else {
    kv = kr[lowBase + d];
    qv = qr[lowBase + d];
  }
  Kb[outBase + 64 + d] = (bf16)kv;
  Qb[outBase + 64 + d] = (bf16)qv;
  krope[idx] = kv;
}

// ---------------- V transpose: (B,S,nh,dh) -> (B,nh,dh,S) ----------------
__global__ __launch_bounds__(256) void transpose_v_kernel(const bf16* __restrict__ v,
                                                          bf16* __restrict__ vt) {
  long long idx = (long long)blockIdx.x * 256 + threadIdx.x;
  if (idx >= 8388608LL) return;
  const int s = (int)(idx & 1023);
  long long r = idx >> 10;
  const int d  = (int)(r & 127);
  const int bh = (int)(r >> 7);
  const int h = bh & 15, b = bh >> 4;
  vt[idx] = v[((long long)(b * 1024 + s)) * 2048 + h * 128 + d];
}

// ---------------- fused attention: O = (K Q^T / sqrt(dh)) V per (b,h) ----------------
// scores[b,h,i,j] = sum_d K[b,i,h,d]*Q[b,j,h,d]; out[b,i,h,:] = scores @ V.
// Block: 128 i-rows for one (b,h). j loop over S in 128 chunks, S-tile staged in LDS.
__global__ __launch_bounds__(256) void attn_fused_kernel(const bf16* __restrict__ Kb,
                                                         const bf16* __restrict__ Qb,
                                                         const bf16* __restrict__ Vt,
                                                         bf16* __restrict__ AO) {
  __shared__ bf16 Ks[128 * 136];
  __shared__ bf16 Qs[128 * 136];
  __shared__ bf16 Vs[128 * 136];   // Vt tile: [d][j], contraction j contiguous
  __shared__ bf16 St[128 * 136];   // scores tile: [i][j]

  const int tid  = threadIdx.x;
  const int lane = tid & 31;
  const int wid  = tid >> 5;
  const int m0   = (wid & 3) * 32;
  const int n0   = (wid >> 2) * 64;
  const int ib   = blockIdx.x * 128;
  const int h    = blockIdx.y;
  const int b    = blockIdx.z;
  const long long tokBase = (long long)b * 1024;
  const float scale = 0.088388347648318447f;   // 1/sqrt(128)

  // K tile resident for the whole block
  for (int i = tid; i < 128 * 16; i += 256) {
    int row = i >> 4, ch = (i & 15) * 8;
    copy16_g2l(Kb + (tokBase + ib + row) * 2048 + h * 128 + ch,
               &Ks[row * 136 + ch]);
  }
  async_join();

  v8f acc2[2][4];
#pragma unroll
  for (int i = 0; i < 2; ++i)
#pragma unroll
    for (int j = 0; j < 4; ++j) acc2[i][j] = vzero();

  const int col = lane & 15;
  const int hi  = lane >> 4;

  for (int jb = 0; jb < 1024; jb += 128) {
    __syncthreads();   // prev iter done with Qs/Vs/St (and Ks fill visible)
    for (int i = tid; i < 128 * 16; i += 256) {
      int row = i >> 4, ch = (i & 15) * 8;
      copy16_g2l(Qb + (tokBase + jb + row) * 2048 + h * 128 + ch,
                 &Qs[row * 136 + ch]);
      copy16_g2l(Vt + ((long long)(b * 16 + h) * 128 + row) * 1024 + jb + ch,
                 &Vs[row * 136 + ch]);
    }
    async_join();
    __syncthreads();

    // stage 1: S = K_i @ Q_j^T  (contraction over d = 128)
    v8f sacc[2][4];
#pragma unroll
    for (int i = 0; i < 2; ++i)
#pragma unroll
      for (int j = 0; j < 4; ++j) sacc[i][j] = vzero();
    for (int k0 = 0; k0 < 128; k0 += 32) {
      v16bf a0 = load_a_frag(Ks, 136, m0, k0, lane);
      v16bf a1 = load_a_frag(Ks, 136, m0 + 16, k0, lane);
#pragma unroll
      for (int j = 0; j < 4; ++j) {
        v16bf bq = load_b_frag(Qs, 136, n0 + 16 * j, k0, lane);
        sacc[0][j] = wmma_bf16(a0, bq, sacc[0][j]);
        sacc[1][j] = wmma_bf16(a1, bq, sacc[1][j]);
      }
    }
    // scale + stash S tile as bf16 for second WMMA
#pragma unroll
    for (int i = 0; i < 2; ++i)
#pragma unroll
      for (int j = 0; j < 4; ++j)
#pragma unroll
        for (int e = 0; e < 8; ++e)
          St[(m0 + 16 * i + 8 * hi + e) * 136 + n0 + 16 * j + col] =
              (bf16)(sacc[i][j][e] * scale);
    __syncthreads();

    // stage 2: O += S @ V_j  (contraction over j = 128; B cols = d via Vt rows)
    for (int k0 = 0; k0 < 128; k0 += 32) {
      v16bf a0 = load_a_frag(St, 136, m0, k0, lane);
      v16bf a1 = load_a_frag(St, 136, m0 + 16, k0, lane);
#pragma unroll
      for (int j = 0; j < 4; ++j) {
        v16bf bv = load_b_frag(Vs, 136, n0 + 16 * j, k0, lane);
        acc2[0][j] = wmma_bf16(a0, bv, acc2[0][j]);
        acc2[1][j] = wmma_bf16(a1, bv, acc2[1][j]);
      }
    }
  }

  // write O (bf16) to attn_out (B,S,nh,dh)
#pragma unroll
  for (int i = 0; i < 2; ++i)
#pragma unroll
    for (int j = 0; j < 4; ++j)
#pragma unroll
      for (int e = 0; e < 8; ++e) {
        long long tok = tokBase + ib + m0 + 16 * i + 8 * hi + e;
        AO[tok * 2048 + h * 128 + n0 + 16 * j + col] = (bf16)acc2[i][j][e];
      }
}

// ---------------- host launcher ----------------
extern "C" void kernel_launch(void* const* d_in, const int* in_sizes, int n_in,
                              void* d_out, int out_size, void* d_ws, size_t ws_size,
                              hipStream_t stream) {
  (void)in_sizes; (void)n_in; (void)out_size; (void)ws_size;
  const float* x     = (const float*)d_in[0];
  const float* W_dkv = (const float*)d_in[1];
  const float* W_dq  = (const float*)d_in[2];
  const float* W_uk  = (const float*)d_in[3];
  const float* W_uv  = (const float*)d_in[4];
  const float* W_uq  = (const float*)d_in[5];
  const float* W_kr  = (const float*)d_in[6];
  const float* W_qr  = (const float*)d_in[7];
  const float* W_out = (const float*)d_in[8];

  float* out_output = (float*)d_out;                    // (B,S,2048)
  float* out_ckv    = out_output + 8388608LL;           // (B,S,512)
  float* out_krope  = out_ckv + 2097152LL;              // (B,S,16,64)

  char* ws = (char*)d_ws;
  size_t off = 0;
  auto take = [&](size_t bytes) -> void* {
    void* p = ws + off;
    off += (bytes + 255) & ~(size_t)255;
    return p;
  };
  bf16* xb     = (bf16*)take(8388608LL * 2);
  bf16* wdkv_t = (bf16*)take(1048576LL * 2);
  bf16* wdq_t  = (bf16*)take(1048576LL * 2);
  bf16* wuk_t  = (bf16*)take(524288LL * 2);
  bf16* wuv_t  = (bf16*)take(1048576LL * 2);
  bf16* wuq_t  = (bf16*)take(524288LL * 2);
  bf16* wkr_t  = (bf16*)take(2097152LL * 2);
  bf16* wqr_t  = (bf16*)take(524288LL * 2);
  bf16* wout_t = (bf16*)take(4194304LL * 2);
  bf16* ckv_b  = (bf16*)take(2097152LL * 2);
  bf16* cq_b   = (bf16*)take(2097152LL * 2);
  bf16* kc_b   = (bf16*)take(4194304LL * 2);
  bf16* qc_b   = (bf16*)take(4194304LL * 2);
  bf16* v_b    = (bf16*)take(8388608LL * 2);
  float* kr_f  = (float*)take(4194304LL * 4);
  float* qr_f  = (float*)take(4194304LL * 4);
  bf16* Kb     = (bf16*)take(8388608LL * 2);
  bf16* Qb     = (bf16*)take(8388608LL * 2);
  bf16* vt     = (bf16*)take(8388608LL * 2);
  bf16* ao_b   = (bf16*)take(8388608LL * 2);

  const int M = 4096;  // B*S
  auto gblk = [](long long n) { return (unsigned)((n + 255) / 256); };

  // precision prep
  cvt_f32_bf16<<<gblk(8388608), 256, 0, stream>>>(x, xb, 8388608LL);
  transpose_cvt<<<gblk(2048LL * 512), 256, 0, stream>>>(W_dkv, wdkv_t, 2048, 512);
  transpose_cvt<<<gblk(2048LL * 512), 256, 0, stream>>>(W_dq, wdq_t, 2048, 512);
  transpose_cvt<<<gblk(512LL * 1024), 256, 0, stream>>>(W_uk, wuk_t, 512, 1024);
  transpose_cvt<<<gblk(512LL * 2048), 256, 0, stream>>>(W_uv, wuv_t, 512, 2048);
  transpose_cvt<<<gblk(512LL * 1024), 256, 0, stream>>>(W_uq, wuq_t, 512, 1024);
  transpose_cvt<<<gblk(2048LL * 1024), 256, 0, stream>>>(W_kr, wkr_t, 2048, 1024);
  transpose_cvt<<<gblk(512LL * 1024), 256, 0, stream>>>(W_qr, wqr_t, 512, 1024);
  transpose_cvt<<<gblk(2048LL * 2048), 256, 0, stream>>>(W_out, wout_t, 2048, 2048);

  // down/up projections
  gemm_bf16_wmma<true, true><<<dim3(512 / 128, M / 128), 256, 0, stream>>>(
      xb, wdkv_t, out_ckv, ckv_b, M, 512, 2048);                 // c_kv (+bf16 copy)
  gemm_bf16_wmma<false, true><<<dim3(512 / 128, M / 128), 256, 0, stream>>>(
      xb, wdq_t, nullptr, cq_b, M, 512, 2048);                   // c_q
  gemm_bf16_wmma<false, true><<<dim3(1024 / 128, M / 128), 256, 0, stream>>>(
      ckv_b, wuk_t, nullptr, kc_b, M, 1024, 512);                // k_c
  gemm_bf16_wmma<false, true><<<dim3(2048 / 128, M / 128), 256, 0, stream>>>(
      ckv_b, wuv_t, nullptr, v_b, M, 2048, 512);                 // v
  gemm_bf16_wmma<false, true><<<dim3(1024 / 128, M / 128), 256, 0, stream>>>(
      cq_b, wuq_t, nullptr, qc_b, M, 1024, 512);                 // q_c
  gemm_bf16_wmma<true, false><<<dim3(1024 / 128, M / 128), 256, 0, stream>>>(
      xb, wkr_t, kr_f, nullptr, M, 1024, 2048);                  // k_r (fp32)
  gemm_bf16_wmma<true, false><<<dim3(1024 / 128, M / 128), 256, 0, stream>>>(
      cq_b, wqr_t, qr_f, nullptr, M, 1024, 512);                 // q_r (fp32)

  // RoPE + pack + k_rope output, V transpose
  rope_pack_kernel<<<gblk(4194304), 256, 0, stream>>>(kr_f, qr_f, kc_b, qc_b,
                                                      Kb, Qb, out_krope);
  transpose_v_kernel<<<gblk(8388608), 256, 0, stream>>>(v_b, vt);

  // fused attention (no softmax in reference)
  attn_fused_kernel<<<dim3(8, 16, 4), 256, 0, stream>>>(Kb, Qb, vt, ao_b);

  // final projection
  gemm_bf16_wmma<true, false><<<dim3(2048 / 128, M / 128), 256, 0, stream>>>(
      ao_b, wout_t, out_output, nullptr, M, 2048, 2048);
}
